// PointnetFPModule_68195490726017
// MI455X (gfx1250) — matrix operand order repaired and checked by reading, not compiled, not run on hardware
//
#include <hip/hip_runtime.h>
#include <hip/hip_bf16.h>

// ---------------------------------------------------------------------------
// Problem constants (from reference setup_inputs):
//   B=8, N=4096 (unknown), M=1024 (known), C1=256, C2=256, Cin=512, H=256
// ---------------------------------------------------------------------------
#define BB    8
#define NN    4096
#define MM    1024
#define C1C   256
#define C2C   256
#define CIN   512
#define HID   256
#define ROWS  (BB * NN)          // 32768 "GEMM rows"
#define EPSF  1e-8f

#if __has_builtin(__builtin_amdgcn_sched_barrier)
#define SCHED_FENCE() __builtin_amdgcn_sched_barrier(0)
#else
#define SCHED_FENCE()
#endif

typedef __attribute__((ext_vector_type(16))) __bf16 v16bf;
typedef __attribute__((ext_vector_type(8)))  float  v8f;

union Frag16 {                   // 16 bf16 = 8 VGPRs, loaded as 2x b128
    v16bf v;
    uint4 u[2];
};

// ---------------------------------------------------------------------------
// Kernel 1: convert + transpose weights to bf16, N-major (so a WMMA B-fragment
// for column n is a contiguous 32B K-run: matches 16-bit B VGPR layout).
//   Wt1[n*512 + k] = bf16(W1[k*256 + n]),  Wt2[n*256 + k] = bf16(W2[k*256 + n])
// ---------------------------------------------------------------------------
__global__ void prep_weights(const float* __restrict__ W1,
                             const float* __restrict__ W2,
                             __bf16* __restrict__ Wt1,
                             __bf16* __restrict__ Wt2) {
    int t = blockIdx.x * blockDim.x + threadIdx.x;
    if (t < CIN * HID) {
        int k = t / HID, n = t % HID;
        Wt1[n * CIN + k] = (__bf16)W1[t];
    } else {
        int u = t - CIN * HID;
        if (u < HID * HID) {
            int k = u / HID, n = u % HID;
            Wt2[n * HID + k] = (__bf16)W2[u];
        }
    }
}

// ---------------------------------------------------------------------------
// Kernel 2: 3-NN (squared distances) + inverse-distance weights.
// One block = 256 query points of one batch; known xyz staged in LDS (12 KB).
// ---------------------------------------------------------------------------
__global__ void three_nn_kernel(const float* __restrict__ unknown, // B*N*3
                                const float* __restrict__ known,   // B*M*3
                                int*   __restrict__ idx3,          // ROWS*3
                                float* __restrict__ w3) {          // ROWS*3
    __shared__ float kx[MM], ky[MM], kz[MM];
    const int blocksPerBatch = NN / 256;                // 16
    const int b     = blockIdx.x / blocksPerBatch;
    const int pBase = (blockIdx.x % blocksPerBatch) * 256;

    const float* kb = known + (size_t)b * MM * 3;
    for (int i = threadIdx.x; i < MM; i += 256) {
        kx[i] = kb[i * 3 + 0];
        ky[i] = kb[i * 3 + 1];
        kz[i] = kb[i * 3 + 2];
    }
    __syncthreads();

    const int p = b * NN + pBase + threadIdx.x;         // global point id
    const float ux = unknown[p * 3 + 0];
    const float uy = unknown[p * 3 + 1];
    const float uz = unknown[p * 3 + 2];

    float d0 = 3.4e38f, d1 = 3.4e38f, d2 = 3.4e38f;
    int   i0 = 0, i1 = 0, i2 = 0;
    for (int j = 0; j < MM; ++j) {
        float dx = ux - kx[j];
        float dy = uy - ky[j];
        float dz = uz - kz[j];
        float d  = dx * dx + dy * dy + dz * dz;
        if (d < d0)      { d2 = d1; i2 = i1; d1 = d0; i1 = i0; d0 = d; i0 = j; }
        else if (d < d1) { d2 = d1; i2 = i1; d1 = d;  i1 = j; }
        else if (d < d2) { d2 = d;  i2 = j; }
    }
    const float r0 = 1.0f / (d0 + EPSF);
    const float r1 = 1.0f / (d1 + EPSF);
    const float r2 = 1.0f / (d2 + EPSF);
    const float rs = 1.0f / (r0 + r1 + r2);
    idx3[p * 3 + 0] = i0;  w3[p * 3 + 0] = r0 * rs;
    idx3[p * 3 + 1] = i1;  w3[p * 3 + 1] = r1 * rs;
    idx3[p * 3 + 2] = i2;  w3[p * 3 + 2] = r2 * rs;
}

// ---------------------------------------------------------------------------
// Kernel 3: interpolation + concat. One block per point, 256 threads = channel.
// Single pass writes BOTH the f32 prop_features output and the bf16 GEMM input.
// ---------------------------------------------------------------------------
__global__ void interp_concat_kernel(const float* __restrict__ known_feats,  // B*M*C2
                                     const float* __restrict__ unknow_feats, // B*N*C1
                                     const int*   __restrict__ idx3,
                                     const float* __restrict__ w3,
                                     float*  __restrict__ prop_out,          // ROWS*512 (d_out region)
                                     __bf16* __restrict__ prop_bf) {         // ROWS*512 (ws)
    const int p = blockIdx.x;               // 0..32767
    const int b = p >> 12;                  // p / 4096
    const int c = threadIdx.x;              // 0..255
    const int i0 = idx3[p * 3 + 0];
    const int i1 = idx3[p * 3 + 1];
    const int i2 = idx3[p * 3 + 2];
    const float w0 = w3[p * 3 + 0];
    const float w1 = w3[p * 3 + 1];
    const float w2 = w3[p * 3 + 2];

    const float* kf = known_feats + (size_t)b * MM * C2C;
    const float interp = w0 * kf[i0 * C2C + c]
                       + w1 * kf[i1 * C2C + c]
                       + w2 * kf[i2 * C2C + c];
    const float skip = unknow_feats[(size_t)p * C1C + c];

    const size_t base = (size_t)p * CIN;
    prop_out[base + c]        = interp;     // interpolated first (reference order)
    prop_out[base + C1C + c]  = skip;
    prop_bf [base + c]        = (__bf16)interp;
    prop_bf [base + C1C + c]  = (__bf16)skip;
}

// ---------------------------------------------------------------------------
// Kernel 4/5: WMMA bf16 GEMM + bias + ReLU, software-pipelined with explicit
// scheduling fences so the prefetch group stays ahead of the WMMA group.
//   A  : ROWS x KTOT  bf16, row-major
//   Bt : 256  x KTOT  bf16, N-major (transposed weights)
//   C  : ROWS x 256   (bf16 when OUT_BF16, else f32)
// Block = 256 threads = 8 waves. Wave w computes a 16(M) x 64(N) strip:
// block tile = 128(M) x 64(N). K-loop in steps of 32.
//
// Pipeline shape per K-step (enforced by __builtin_amdgcn_sched_barrier):
//   [ 10 global_load_b128 for step s+1 ] || [ 4 v_wmma for step s ]
// so the wait ahead of a WMMA group only drains loads issued a full step
// earlier (s_wait_loadcnt ~10), overlapping L2 latency with matrix math.
//
// Fragment loads follow the CDNA5 16-bit VGPR layouts exactly:
//   A (16x32): lane L -> M = L%16, h = L/16; K-runs [h*8, h*8+8) and
//              [16+h*8, 16+h*8+8)  -> two contiguous b128 loads.
//   B (32x16): lane L -> N = L%16, h = L/16; K-runs [h*8, h*8+8) and
//              [16+h*8, 16+h*8+8) -> two contiguous b128 loads from N-major Bt.
//   C/D (16x16 f32): VGPR r -> row h*8+r, col = L%16.
// ---------------------------------------------------------------------------
template <int KTOT, bool OUT_BF16>
__global__ void wmma_gemm_bias_relu(const __bf16* __restrict__ A,
                                    const __bf16* __restrict__ Bt,
                                    const float*  __restrict__ bias,
                                    void* __restrict__ Cout) {
    constexpr int KSTEPS = KTOT / 32;

    const int lane  = threadIdx.x & 31;
    const int wave  = threadIdx.x >> 5;
    const int h     = lane >> 4;        // half-wave id
    const int nl    = lane & 15;
    const int mBase = blockIdx.x * 128 + wave * 16;
    const int nBase = blockIdx.y * 64;
    const int aRow  = mBase + nl;       // A-fragment row for this lane

    // Per-lane base pointers (K-offset folded in per step).
    const __bf16* ap  = A + (size_t)aRow * KTOT + h * 8;
    const __bf16* bp0 = Bt + (size_t)(nBase + 0 * 16 + nl) * KTOT + h * 8;
    const __bf16* bp1 = Bt + (size_t)(nBase + 1 * 16 + nl) * KTOT + h * 8;
    const __bf16* bp2 = Bt + (size_t)(nBase + 2 * 16 + nl) * KTOT + h * 8;
    const __bf16* bp3 = Bt + (size_t)(nBase + 3 * 16 + nl) * KTOT + h * 8;
    const __bf16* bps[4] = { bp0, bp1, bp2, bp3 };

#if __has_builtin(__builtin_prefetch)
    __builtin_prefetch(ap, 0, 3);       // global_prefetch_b8: warm first line
#endif

    v8f acc[4] = {};

    Frag16 aF[2];
    Frag16 bF[2][4];

    // Prologue: load K-step 0 into buffer 0.
    aF[0].u[0] = *(const uint4*)(ap);        // K = h*8      .. +7
    aF[0].u[1] = *(const uint4*)(ap + 16);   // K = 16+h*8   .. +7
#pragma unroll
    for (int t = 0; t < 4; ++t) {
        bF[0][t].u[0] = *(const uint4*)(bps[t]);      // K = h*8      .. +7
        bF[0][t].u[1] = *(const uint4*)(bps[t] + 16); // K = 16+h*8   .. +7
    }

#pragma unroll
    for (int s = 0; s < KSTEPS; ++s) {
        const int cur = s & 1;
        const int nxt = cur ^ 1;

        // Prefetch K-step s+1 into the alternate buffer.
        if (s + 1 < KSTEPS) {
            const int kk = (s + 1) * 32;
            aF[nxt].u[0] = *(const uint4*)(ap + kk);
            aF[nxt].u[1] = *(const uint4*)(ap + kk + 16);
#pragma unroll
            for (int t = 0; t < 4; ++t) {
                bF[nxt][t].u[0] = *(const uint4*)(bps[t] + kk);
                bF[nxt][t].u[1] = *(const uint4*)(bps[t] + kk + 16);
            }
        }

        // Keep the prefetch group ahead of the WMMA group: nothing may cross.
        SCHED_FENCE();

        // Four back-to-back WMMAs sharing one A fragment.
#pragma unroll
        for (int t = 0; t < 4; ++t) {
            acc[t] = __builtin_amdgcn_wmma_f32_16x16x32_bf16(
                false, aF[cur].v, false, bF[cur][t].v, (short)0, acc[t],
                false, false);
        }

        SCHED_FENCE();
    }

    // Epilogue: bias + ReLU, store per documented C/D layout.
#pragma unroll
    for (int t = 0; t < 4; ++t) {
        const int col = nBase + t * 16 + nl;
        const float bv = bias[col];
#pragma unroll
        for (int r = 0; r < 8; ++r) {
            const int row = mBase + h * 8 + r;
            float v = acc[t][r] + bv;
            v = v > 0.0f ? v : 0.0f;
            if (OUT_BF16)
                ((__bf16*)Cout)[(size_t)row * 256 + col] = (__bf16)v;
            else
                ((float*)Cout)[(size_t)row * 256 + col] = v;
        }
    }
}

// ---------------------------------------------------------------------------
// Host-side launch. Workspace layout (bytes, all 256B-aligned):
//   idx3    @ 0         : 32768*3*4  =   393,216
//   w3      @ 393216    : 32768*3*4  =   393,216
//   Wt1     @ 786432    : 512*256*2  =   262,144
//   Wt2     @ 1048576   : 256*256*2  =   131,072
//   prop_bf @ 1179648   : 32768*512*2 = 33,554,432
//   h_bf    @ 34734080  : 32768*256*2 = 16,777,216
//   total ~ 49.2 MB
// d_out: out f32 (32768*256) then prop_features f32 (32768*512).
// ---------------------------------------------------------------------------
extern "C" void kernel_launch(void* const* d_in, const int* in_sizes, int n_in,
                              void* d_out, int out_size, void* d_ws, size_t ws_size,
                              hipStream_t stream) {
    const float* unknown      = (const float*)d_in[0];
    const float* known        = (const float*)d_in[1];
    const float* unknow_feats = (const float*)d_in[2];
    const float* known_feats  = (const float*)d_in[3];
    const float* W1           = (const float*)d_in[4];
    const float* b1           = (const float*)d_in[5];
    const float* W2           = (const float*)d_in[6];
    const float* b2           = (const float*)d_in[7];

    char* ws = (char*)d_ws;
    int*    idx3    = (int*)   (ws + 0);
    float*  w3      = (float*) (ws + 393216);
    __bf16* Wt1     = (__bf16*)(ws + 786432);
    __bf16* Wt2     = (__bf16*)(ws + 1048576);
    __bf16* prop_bf = (__bf16*)(ws + 1179648);
    __bf16* h_bf    = (__bf16*)(ws + 34734080ull);

    float* out_f32   = (float*)d_out;                       // ROWS*256
    float* prop_out  = out_f32 + (size_t)ROWS * HID;        // ROWS*512

    // 1) weights -> bf16, transposed (N-major)
    {
        int total = CIN * HID + HID * HID;                  // 196608
        prep_weights<<<(total + 255) / 256, 256, 0, stream>>>(W1, W2, Wt1, Wt2);
    }
    // 2) 3-NN + weights
    three_nn_kernel<<<BB * (NN / 256), 256, 0, stream>>>(unknown, known, idx3, w3);
    // 3) interpolate + concat (writes prop_features output and bf16 GEMM input)
    interp_concat_kernel<<<ROWS, 256, 0, stream>>>(known_feats, unknow_feats,
                                                   idx3, w3, prop_out, prop_bf);
    // 4) h = relu(prop @ W1 + b1)  -> bf16
    {
        dim3 grid(ROWS / 128, HID / 64);                    // (256, 4)
        wmma_gemm_bias_relu<CIN, true><<<grid, 256, 0, stream>>>(prop_bf, Wt1, b1, h_bf);
    }
    // 5) out = relu(h @ W2 + b2)   -> f32
    {
        dim3 grid(ROWS / 128, HID / 64);                    // (256, 4)
        wmma_gemm_bias_relu<HID, false><<<grid, 256, 0, stream>>>(h_bf, Wt2, b2, out_f32);
    }
}
